// SHARElayer_14499809591707
// MI455X (gfx1250) — matrix operand order repaired and checked by reading, not compiled, not run on hardware
//
#include <hip/hip_runtime.h>
#include <math.h>

// ---------------------------------------------------------------------------
// SHARE hypergraph-attention layer for MI455X (gfx1250, wave32, WMMA + TDM).
// N=20000 nodes, E=4096 hyperedges, D_IN=256, D_OUT=128.
// Big GEMMs: bf16 v_wmma_f32_16x16x32_bf16 with f32 accumulation.
// k_node stages its per-chunk operand tiles through the Tensor Data Mover
// (tensor_load_to_lds + s_wait_tensorcnt) — 6-arg clang-23 builtin form.
// ---------------------------------------------------------------------------

typedef __attribute__((ext_vector_type(16))) __bf16 v16bf;
typedef __attribute__((ext_vector_type(8)))  __bf16 v8bf;
typedef __attribute__((ext_vector_type(8)))  float  v8f;
typedef __attribute__((ext_vector_type(4)))  unsigned v4u;
typedef __attribute__((ext_vector_type(8)))  int      v8i_t;
typedef __attribute__((ext_vector_type(4)))  int      v4i_t;

static constexpr int Nn   = 20000;
static constexpr int Ee   = 4096;
static constexpr int Din  = 256;
static constexpr int Dout = 128;
static constexpr float INVD = 0.08838834764831845f;   // 1/sqrt(128)

#if defined(__gfx1250__) && __has_builtin(__builtin_amdgcn_tensor_load_to_lds) && \
    __has_builtin(__builtin_amdgcn_s_wait_tensorcnt)
#define USE_TDM 1
#else
#define USE_TDM 0
#endif

static __device__ __forceinline__ __bf16 f2bf(float f) {
    unsigned u = __float_as_uint(f);
    unsigned r = (u + 0x7FFFu + ((u >> 16) & 1u)) >> 16;
    unsigned short s = (unsigned short)r;
    __bf16 b;
    __builtin_memcpy(&b, &s, 2);
    return b;
}

static __device__ __forceinline__ v16bf cat8(v8bf lo, v8bf hi) {
    return __builtin_shufflevector(lo, hi, 0,1,2,3,4,5,6,7,8,9,10,11,12,13,14,15);
}

static __device__ __forceinline__ v8f v8f_zero() {
    v8f z;
#pragma unroll
    for (int i = 0; i < 8; ++i) z[i] = 0.0f;
    return z;
}

static __device__ __forceinline__ v8f wmma_bf16(v16bf a, v16bf b, v8f c) {
    return __builtin_amdgcn_wmma_f32_16x16x32_bf16(false, a, false, b, (short)0, c,
                                                   false, false);
}

#if USE_TDM
// Workgroup-relative LDS byte offset of a __shared__ pointer.
static __device__ __forceinline__ unsigned lds_off(void* p) {
    return (unsigned)(unsigned long long)(__attribute__((address_space(3))) char*)p;
}

// 2D tile load via Tensor Data Mover. Element size fixed at 2 bytes (bf16).
// pad_int/pad_amt are the encoded D# fields (pad every 2^(pad_int+1) dwords by
// (pad_amt+1) dwords) so padded LDS row strides stay bank-friendly.
static __device__ __forceinline__ void tdm_load_2d(
    unsigned ldsaddr, const void* gptr,
    unsigned tensor_d0, unsigned tensor_d1,
    unsigned tile_d0, unsigned tile_d1,
    unsigned stride0, unsigned pad_int, unsigned pad_amt) {
    unsigned long long ga = (unsigned long long)gptr;
    v4u g0;
    g0[0] = 1u;                                          // count=1, user mode
    g0[1] = ldsaddr;                                     // lds_addr (bytes)
    g0[2] = (unsigned)(ga & 0xFFFFFFFFu);                // global_addr[31:0]
    g0[3] = (unsigned)((ga >> 32) & 0x01FFFFFFu) | (2u << 30);  // addr hi | type=2
    v8i_t g1;
    unsigned w0 = (1u << 16)                              // data_size = 2 bytes
                | (1u << 20)                              // pad_enable
                | (pad_int << 22) | (pad_amt << 25);
    g1[0] = (int)w0;
    g1[1] = (int)((tensor_d0 & 0xFFFFu) << 16);           // barrier_addr=0 | dim0.lo
    g1[2] = (int)(((tensor_d0 >> 16) & 0xFFFFu) | ((tensor_d1 & 0xFFFFu) << 16));
    g1[3] = (int)(((tensor_d1 >> 16) & 0xFFFFu) | ((tile_d0 & 0xFFFFu) << 16));
    g1[4] = (int)(tile_d1 & 0xFFFFu);                     // tile_dim2 = 0
    g1[5] = (int)stride0;                                 // tensor_dim0_stride lo
    g1[6] = 0;
    g1[7] = 0;
    v4i_t gz4;
    gz4[0] = gz4[1] = gz4[2] = gz4[3] = 0;
    v8i_t gz8;
#pragma unroll
    for (int i = 0; i < 8; ++i) gz8[i] = 0;
    __builtin_amdgcn_tensor_load_to_lds(g0, g1, gz4, gz4, gz8, 0);
}
#endif

// ---------------------------------------------------------------------------
// k_prep: w1a = w1 @ a ; zero colsum ; cw2t[n][k] = bf16(clide_w2[k][n])
// ---------------------------------------------------------------------------
__global__ __launch_bounds__(256)
void k_prep(const float* __restrict__ w1, const float* __restrict__ a,
            const float* __restrict__ cw2, float* __restrict__ w1a,
            float* __restrict__ colsum, __bf16* __restrict__ cw2t) {
    int tid = threadIdx.x;
    float s = 0.0f;
#pragma unroll 8
    for (int k = 0; k < Dout; ++k) s += w1[tid * Dout + k] * a[k];
    w1a[tid] = s;
    for (int j = tid; j < Ee; j += 256) colsum[j] = 0.0f;
    for (int idx = tid; idx < Dout * Dout; idx += 256) {
        int n = idx >> 7, k = idx & 127;
        cw2t[n * Dout + k] = f2bf(cw2[k * Dout + n]);
    }
}

// ---------------------------------------------------------------------------
// k_scores: p[n] = exp(8*tanh((x[n]·w1a)/(8*sqrt(D)))) ; xb = bf16(x)
// ---------------------------------------------------------------------------
__global__ __launch_bounds__(256)
void k_scores(const float* __restrict__ x, const float* __restrict__ w1a,
              float* __restrict__ p, __bf16* __restrict__ xb) {
    int lane = threadIdx.x & 31;
    int node = blockIdx.x * 8 + (threadIdx.x >> 5);
    size_t base = (size_t)node * Din;
    float s = 0.0f;
#pragma unroll
    for (int i = 0; i < 8; ++i) {
        int d = lane + 32 * i;
        float v = x[base + d];
        s += v * w1a[d];
        xb[base + d] = f2bf(v);
    }
#pragma unroll
    for (int off = 16; off; off >>= 1) s += __shfl_xor(s, off, 32);
    if (lane == 0) p[node] = __expf(8.0f * tanhf(s * (INVD * 0.125f)));
}

// ---------------------------------------------------------------------------
// k_colsum: colsum[j] += sum_n p[n]*H[n,j]  (grid.y partitions N)
// ---------------------------------------------------------------------------
__global__ __launch_bounds__(256)
void k_colsum(const float* __restrict__ H, const float* __restrict__ p,
              float* __restrict__ colsum) {
    int j = blockIdx.x * 256 + threadIdx.x;
    int nbeg = blockIdx.y * 1000;
    float acc = 0.0f;
    for (int i = 0; i < 1000; ++i) {
        float pv = p[nbeg + i];                       // block-uniform -> SGPR
        acc += pv * H[(size_t)(nbeg + i) * Ee + j];
    }
    atomicAdd(&colsum[j], acc);
}

// ---------------------------------------------------------------------------
// k_xw3: xw3b = bf16(x @ w3)   [N,128]
// ---------------------------------------------------------------------------
__global__ __launch_bounds__(256)
void k_xw3(const float* __restrict__ x, const float* __restrict__ w3,
           __bf16* __restrict__ xw3b) {
    __shared__ __align__(16) float xs[32 * 260];
    int tid = threadIdx.x;
    int n0 = blockIdx.x * 32;
    {
        int rn = tid >> 3, c0 = (tid & 7) * 32;
        const float4* src = (const float4*)(x + (size_t)(n0 + rn) * Din + c0);
#pragma unroll
        for (int i = 0; i < 8; ++i)
            *(float4*)&xs[rn * 260 + c0 + i * 4] = src[i];
    }
    __syncthreads();
    int m = tid >> 3, d0 = (tid & 7) * 16;
#pragma unroll
    for (int dd = 0; dd < 16; ++dd) {
        int d = d0 + dd;
        float acc = 0.0f;
        for (int k = 0; k < Din; ++k) acc += xs[m * 260 + k] * w3[k * Dout + d];
        xw3b[(size_t)(n0 + m) * Dout + d] = f2bf(acc);
    }
}

// ---------------------------------------------------------------------------
// k_edge_feat: EFp[slab][j][d] = sum_{n in slab} p[n]*H[n,j]*x[n,d]
// WMMA bf16. A = (p*H)^T, weighted+transposed into LDS on the fly.
// B = xb chunk, staged TRANSPOSED in LDS so each B fragment is two
// ds_load_b128 (no scalar gather / v_mov_b16 assembly).
// ---------------------------------------------------------------------------
__global__ __launch_bounds__(256)
void k_edge_feat(const float* __restrict__ H, const __bf16* __restrict__ xb,
                 const float* __restrict__ p, float* __restrict__ EFp) {
    __shared__ __align__(16) __bf16 Ab[64 * 40];     // [j_local][n_local]
    __shared__ __align__(16) __bf16 BbT[256 * 40];   // [d][n_local]
    const int tid  = threadIdx.x;
    const int lane = tid & 31;
    const int widx = tid >> 5;
    const int jt   = widx & 3;        // j-tile within block (4)
    const int dgrp = widx >> 2;       // d half (2 x 128)
    const int j0   = blockIdx.x * 64;
    const int slab = blockIdx.y;
    const int kbeg = slab * 157;
    const int kend = (kbeg + 157 < 625) ? kbeg + 157 : 625;

    v8f acc[8];
#pragma unroll
    for (int t = 0; t < 8; ++t) acc[t] = v8f_zero();

    const int m     = lane & 15;
    const int kb8   = (lane >> 4) * 8;
    const int kb16  = (lane >> 4) * 16;
    const int col16 = lane & 15;

    for (int ks = kbeg; ks < kend; ++ks) {
        const int n0k = ks * 32;
        __syncthreads();
        {   // stage weighted H^T chunk: 32 n x 64 j -> Ab[j][n]
            int rn = tid >> 3, c0 = (tid & 7) * 8;
            float pv = p[n0k + rn];
            const float4* Hp = (const float4*)(H + (size_t)(n0k + rn) * Ee + j0 + c0);
            float4 h0 = Hp[0], h1 = Hp[1];
            Ab[(c0 + 0) * 40 + rn] = f2bf(h0.x * pv);
            Ab[(c0 + 1) * 40 + rn] = f2bf(h0.y * pv);
            Ab[(c0 + 2) * 40 + rn] = f2bf(h0.z * pv);
            Ab[(c0 + 3) * 40 + rn] = f2bf(h0.w * pv);
            Ab[(c0 + 4) * 40 + rn] = f2bf(h1.x * pv);
            Ab[(c0 + 5) * 40 + rn] = f2bf(h1.y * pv);
            Ab[(c0 + 6) * 40 + rn] = f2bf(h1.z * pv);
            Ab[(c0 + 7) * 40 + rn] = f2bf(h1.w * pv);
        }
        {   // stage xb chunk transposed: 32 n x 256 d -> BbT[d][n]
            int rn = tid >> 3, c0 = (tid & 7) * 32;
            const v8bf* xp = (const v8bf*)(xb + (size_t)(n0k + rn) * Din + c0);
#pragma unroll
            for (int i = 0; i < 4; ++i) {
                v8bf v = xp[i];
#pragma unroll
                for (int e = 0; e < 8; ++e)
                    BbT[(c0 + i * 8 + e) * 40 + rn] = v[e];
            }
        }
        __syncthreads();

        // A fragment for this wave's j-tile
        int abase = (jt * 16 + m) * 40;
        v16bf af = cat8(*(const v8bf*)&Ab[abase + kb8],
                        *(const v8bf*)&Ab[abase + 16 + kb8]);

#pragma unroll
        for (int dt = 0; dt < 8; ++dt) {
            int col = dgrp * 128 + dt * 16 + col16;
            const __bf16* src = &BbT[col * 40 + kb8];
            v16bf bf = cat8(*(const v8bf*)src, *(const v8bf*)(src + 16));
            acc[dt] = wmma_bf16(af, bf, acc[dt]);
        }
    }

    const int mrow = 8 * (lane >> 4);
#pragma unroll
    for (int dt = 0; dt < 8; ++dt) {
#pragma unroll
        for (int r = 0; r < 8; ++r) {
            int j = j0 + jt * 16 + r + mrow;
            int d = dgrp * 128 + dt * 16 + (lane & 15);
            EFp[((size_t)slab * Ee + j) * Din + d] = acc[dt][r];
        }
    }
}

// ---------------------------------------------------------------------------
// k_edge_mm: EFn = (sum slabs)/colsum ; ef = EFn@clide_w1 ; ew2 = ef@w2
// emits e_feat^T (bf16, [128][E]) and ew2 (bf16, [E][128]) for k_node.
// ---------------------------------------------------------------------------
__global__ __launch_bounds__(256)
void k_edge_mm(const float* __restrict__ EFp, const float* __restrict__ colsum,
               const float* __restrict__ cw1, const float* __restrict__ w2,
               __bf16* __restrict__ efTb, __bf16* __restrict__ ew2b) {
    __shared__ float efn[32 * 260];
    __shared__ float efs[32 * 132];
    int tid = threadIdx.x;
    int j0 = blockIdx.x * 32;
    {
        int rm = tid >> 3, c0 = (tid & 7) * 32;
        float inv = 1.0f / colsum[j0 + rm];
        for (int c = c0; c < c0 + 32; ++c) {
            float s = 0.0f;
#pragma unroll
            for (int pp = 0; pp < 4; ++pp)
                s += EFp[((size_t)pp * Ee + (j0 + rm)) * Din + c];
            efn[rm * 260 + c] = s * inv;
        }
    }
    __syncthreads();
    int mm = tid >> 3, d0 = (tid & 7) * 16;
#pragma unroll
    for (int dd = 0; dd < 16; ++dd) {
        int d = d0 + dd;
        float acc = 0.0f;
        for (int k = 0; k < Din; ++k) acc += efn[mm * 260 + k] * cw1[k * Dout + d];
        efs[mm * 132 + d] = acc;
        efTb[(size_t)d * Ee + j0 + mm] = f2bf(acc);
    }
    __syncthreads();
#pragma unroll
    for (int dd = 0; dd < 16; ++dd) {
        int d = d0 + dd;
        float acc = 0.0f;
        for (int k = 0; k < Dout; ++k) acc += efs[mm * 132 + k] * w2[k * Dout + d];
        ew2b[(size_t)(j0 + mm) * Dout + d] = f2bf(acc);
    }
}

// ---------------------------------------------------------------------------
// k_node: fused edge->node attention + aggregation + clide_w2 + LeakyReLU.
// One wave owns 16 nodes; E streamed in chunks of 32. Per chunk the block
// stages ew2 (32x128) and e_feat^T (128x32) tiles into LDS once — via the
// Tensor Data Mover (padded LDS rows keep strides bank-clean), shared by
// all 8 waves. Three WMMA stages per chunk/epilogue.
// ---------------------------------------------------------------------------
__global__ __launch_bounds__(256)
void k_node(const float* __restrict__ H, const __bf16* __restrict__ xw3b,
            const __bf16* __restrict__ ew2b, const __bf16* __restrict__ efTb,
            const __bf16* __restrict__ cw2t, float* __restrict__ out) {
    __shared__ __align__(16) __bf16 Tb[8][16 * 40];   // per-wave transpose buf
    __shared__ __align__(16) __bf16 ew2S[32 * 136];   // [j_local][k], 272B rows
    __shared__ __align__(16) __bf16 efS[128 * 40];    // [d][j_local], 80B rows
    const int tid  = threadIdx.x;
    const int lane = tid & 31;
    const int widx = tid >> 5;
    const int gw   = blockIdx.x * 8 + widx;
    const bool active = gw < Nn / 16;
    const int gwc = active ? gw : (Nn / 16 - 1);
    const int n0 = gwc * 16;

    const int m    = lane & 15;
    const int kb8  = (lane >> 4) * 8;
    const int kb16 = (lane >> 4) * 16;
    const int mrow = 8 * (lane >> 4);
    const int ncol = lane & 15;

    // A fragments of xw3 row block (16 nodes x 128), resident all loop long
    v16bf axw[4];
#pragma unroll
    for (int kk = 0; kk < 4; ++kk) {
        const __bf16* src = xw3b + (size_t)(n0 + m) * Dout + kk * 32 + kb8;
        axw[kk] = cat8(*(const v8bf*)src, *(const v8bf*)(src + 16));
    }

    v8f oacc[8];
#pragma unroll
    for (int t = 0; t < 8; ++t) oacc[t] = v8f_zero();
    float rs[8];
#pragma unroll
    for (int r = 0; r < 8; ++r) rs[r] = 0.0f;

    __bf16* tb = &Tb[widx][0];

    for (int jc = 0; jc < Ee / 32; ++jc) {
        const int j0 = jc * 32;
        __syncthreads();                       // protect LDS tiles (WAR)
#if USE_TDM
        if (widx == 0) {
            // ew2 chunk: 32 rows x 128 bf16; pad 4 dwords every 64 dwords
            tdm_load_2d(lds_off(ew2S), ew2b + (size_t)j0 * Dout,
                        Dout, 32, Dout, 32, Dout, 5u, 3u);
            // e_feat^T chunk: 128 rows x 32 bf16; pad 4 dwords every 16 dwords
            tdm_load_2d(lds_off(efS), efTb + j0,
                        32, Dout, 32, Dout, Ee, 3u, 3u);
            __builtin_amdgcn_s_wait_tensorcnt(0);
        }
#else
        {
            int row = tid >> 3, c0 = (tid & 7) * 16;
            const v8bf* s = (const v8bf*)(ew2b + (size_t)(j0 + row) * Dout + c0);
            *(v8bf*)&ew2S[row * 136 + c0] = s[0];
            *(v8bf*)&ew2S[row * 136 + c0 + 8] = s[1];
        }
        {
            int dr = tid >> 1, c0 = (tid & 1) * 16;
            const v8bf* s = (const v8bf*)(efTb + (size_t)dr * Ee + j0 + c0);
            *(v8bf*)&efS[dr * 40 + c0] = s[0];
            *(v8bf*)&efS[dr * 40 + c0 + 8] = s[1];
        }
#endif
        __syncthreads();

        // ---- score GEMM: sc = xw3_tile @ ew2_chunk^T  (two 16x16 tiles)
        v8f sc[2];
#pragma unroll
        for (int st = 0; st < 2; ++st) {
            int jloc = st * 16 + ncol;
            v8f c = v8f_zero();
#pragma unroll
            for (int kk = 0; kk < 4; ++kk) {
                const __bf16* src = &ew2S[jloc * 136 + kk * 32 + kb8];
                v16bf bf = cat8(*(const v8bf*)src, *(const v8bf*)(src + 16));
                c = wmma_bf16(axw[kk], bf, c);
            }
            sc[st] = c;
        }

        // ---- elementwise: tanh/exp/mask, rowsum, write t (bf16) to LDS
#pragma unroll
        for (int st = 0; st < 2; ++st) {
#pragma unroll
            for (int r = 0; r < 8; ++r) {
                int mi = r + mrow;
                float v = sc[st][r] * INVD;
                v = __expf(8.0f * tanhf(v * 0.125f));
                v *= H[(size_t)(n0 + mi) * Ee + j0 + st * 16 + ncol];
                rs[r] += v;
                tb[mi * 40 + st * 16 + ncol] = f2bf(v);
            }
        }
        __builtin_amdgcn_wave_barrier();

        // ---- t as A fragment (16 nodes x 32 edges)
        v16bf at = cat8(*(const v8bf*)&tb[m * 40 + kb8],
                        *(const v8bf*)&tb[m * 40 + 16 + kb8]);

        // ---- aggregation: oacc += t @ e_feat_chunk  (8 d-tiles)
#pragma unroll
        for (int dt = 0; dt < 8; ++dt) {
            int d = dt * 16 + ncol;
            const __bf16* src = &efS[d * 40 + kb8];
            v16bf bf = cat8(*(const v8bf*)src, *(const v8bf*)(src + 16));
            oacc[dt] = wmma_bf16(at, bf, oacc[dt]);
        }
        __builtin_amdgcn_wave_barrier();
    }

    // ---- rowsum reduction across the 16 lanes sharing each m
    float inv[8];
#pragma unroll
    for (int r = 0; r < 8; ++r) {
        float s = rs[r];
#pragma unroll
        for (int off = 1; off < 16; off <<= 1) s += __shfl_xor(s, off, 32);
        inv[r] = 1.0f / s;
    }

    // ---- final GEMM: out = leaky((oacc/rowsum) @ clide_w2), K=128 via WMMA
    v8f facc[8];
#pragma unroll
    for (int t = 0; t < 8; ++t) facc[t] = v8f_zero();

#pragma unroll
    for (int kc = 0; kc < 4; ++kc) {
#pragma unroll
        for (int st = 0; st < 2; ++st) {
            int dt = kc * 2 + st;
#pragma unroll
            for (int r = 0; r < 8; ++r)
                tb[(r + mrow) * 40 + st * 16 + ncol] = f2bf(oacc[dt][r] * inv[r]);
        }
        __builtin_amdgcn_wave_barrier();
        v16bf at = cat8(*(const v8bf*)&tb[m * 40 + kb8],
                        *(const v8bf*)&tb[m * 40 + 16 + kb8]);
#pragma unroll
        for (int dn = 0; dn < 8; ++dn) {
            int d = dn * 16 + ncol;
            const __bf16* src = cw2t + (size_t)d * Dout + kc * 32 + kb8;
            v16bf bf = cat8(*(const v8bf*)src, *(const v8bf*)(src + 16));
            facc[dn] = wmma_bf16(at, bf, facc[dn]);
        }
        __builtin_amdgcn_wave_barrier();
    }

    if (active) {
#pragma unroll
        for (int dn = 0; dn < 8; ++dn) {
#pragma unroll
            for (int r = 0; r < 8; ++r) {
                float v = facc[dn][r];
                v = v > 0.0f ? v : 0.1f * v;
                out[(size_t)(n0 + r + mrow) * Dout + dn * 16 + ncol] = v;
            }
        }
    }
}

// ---------------------------------------------------------------------------
extern "C" void kernel_launch(void* const* d_in, const int* in_sizes, int n_in,
                              void* d_out, int out_size, void* d_ws, size_t ws_size,
                              hipStream_t stream) {
    (void)in_sizes; (void)n_in; (void)out_size; (void)ws_size;
    const float* x   = (const float*)d_in[0];
    const float* H   = (const float*)d_in[1];
    const float* w1  = (const float*)d_in[2];
    const float* cw1 = (const float*)d_in[3];
    const float* w2  = (const float*)d_in[4];
    const float* cw2 = (const float*)d_in[5];
    const float* w3  = (const float*)d_in[6];
    const float* a   = (const float*)d_in[7];
    float* out = (float*)d_out;

    size_t off = 0;
    auto carve = [&](size_t bytes) -> char* {
        char* p = (char*)d_ws + off;
        off += (bytes + 255) & ~(size_t)255;
        return p;
    };
    float*  W1A    = (float*) carve(Din * 4);
    float*  P      = (float*) carve((size_t)Nn * 4);
    float*  COLSUM = (float*) carve((size_t)Ee * 4);
    __bf16* XB     = (__bf16*)carve((size_t)Nn * Din * 2);
    float*  EFP    = (float*) carve((size_t)4 * Ee * Din * 4);
    __bf16* EFTB   = (__bf16*)carve((size_t)Dout * Ee * 2);
    __bf16* EW2B   = (__bf16*)carve((size_t)Ee * Dout * 2);
    __bf16* XW3B   = (__bf16*)carve((size_t)Nn * Dout * 2);
    __bf16* CW2T   = (__bf16*)carve((size_t)Dout * Dout * 2);

    k_prep     <<<1,               256, 0, stream>>>(w1, a, cw2, W1A, COLSUM, CW2T);
    k_scores   <<<Nn / 8,          256, 0, stream>>>(x, W1A, P, XB);
    k_colsum   <<<dim3(16, 20),    256, 0, stream>>>(H, P, COLSUM);
    k_xw3      <<<Nn / 32,         256, 0, stream>>>(x, w3, XW3B);
    k_edge_feat<<<dim3(64, 4),     256, 0, stream>>>(H, XB, P, EFP);
    k_edge_mm  <<<Ee / 32,         256, 0, stream>>>(EFP, COLSUM, cw1, w2, EFTB, EW2B);
    k_node     <<<(Nn / 16 + 7)/8, 256, 0, stream>>>(H, XW3B, EW2B, EFTB, CW2T, out);
}
